// TopKGate_41686952575624
// MI455X (gfx1250) — matrix operand order repaired and checked by reading, not compiled, not run on hardware
//
#include <hip/hip_runtime.h>
#include <hip/hip_bf16.h>
#include <math.h>

typedef __attribute__((ext_vector_type(16))) __bf16 v16bf;
typedef __attribute__((ext_vector_type(8)))  __bf16 v8bf;
typedef __attribute__((ext_vector_type(8)))  float  v8f;

#define D_DIM 4096
#define E_DIM 64
#define KC 64              // K-chunk staged in LDS per iteration (128 B/expert in bf16)
#define WPAD 80            // padded LDS row stride (elements) -> 160 B, 32 B-aligned
#define ROWS_PER_BLOCK 128 // 8 waves x 16 rows

__device__ __forceinline__ __bf16 to_bf16(float f) { return (__bf16)f; }

// ---------------- prep: W f32 -> bf16 into workspace, zero the expert accumulator
__global__ __launch_bounds__(256) void moe_gate_prep(const float* __restrict__ W,
                                                     __bf16* __restrict__ Wb,
                                                     float* __restrict__ acc) {
  const int i = (blockIdx.x * 256 + threadIdx.x) * 8;   // 8 elems/thread
  float4 f0 = *(const float4*)(W + i);
  float4 f1 = *(const float4*)(W + i + 4);
  v8bf o;
  o[0] = to_bf16(f0.x); o[1] = to_bf16(f0.y); o[2] = to_bf16(f0.z); o[3] = to_bf16(f0.w);
  o[4] = to_bf16(f1.x); o[5] = to_bf16(f1.y); o[6] = to_bf16(f1.z); o[7] = to_bf16(f1.w);
  *(v8bf*)(Wb + i) = o;
  if (blockIdx.x == 0 && threadIdx.x < E_DIM) acc[threadIdx.x] = 0.0f;
}

// ---------------- async copy: one 16 B segment of bf16 W into LDS (CDNA5 async-to-LDS path)
__device__ __forceinline__ void async_b128_to_lds(unsigned lds_addr, unsigned long long gaddr) {
  asm volatile("global_load_async_to_lds_b128 %0, %1, off"
               :: "v"(lds_addr), "v"(gaddr) : "memory");
}

__global__ __launch_bounds__(256) void moe_gate_gemm(const float* __restrict__ x,
                                                     const __bf16* __restrict__ Wb,
                                                     float* __restrict__ out,
                                                     float* __restrict__ expert_acc,
                                                     int N) {
  __shared__ __align__(32) __bf16 wlds[2][E_DIM][WPAD];      // 2 x 10.0 KB, double-buffered
  __shared__ float sh_logits[ROWS_PER_BLOCK][E_DIM + 1];     // 32.5 KB, stride 65 vs bank conflicts
  __shared__ float sh_acc[E_DIM];                            // 256 B

  const int tid    = threadIdx.x;
  const int lane   = tid & 31;
  const int wave   = tid >> 5;     // 0..7
  const int laneLo = lane & 15;
  const int hi     = lane >> 4;    // 0 or 1

  const int rowBase = blockIdx.x * ROWS_PER_BLOCK;
  const int myRow   = rowBase + wave * 16 + laneLo;          // A-fragment row for this lane
  const float* xrow = x + (long long)myRow * D_DIM;

  if (tid < E_DIM) sh_acc[tid] = 0.0f;

  // segment decomposition of one W K-chunk: 64 experts x 128 B = 512 x 16 B, 2 per thread
  const int e0 = tid >> 3;                 // expert for segment 0 (0..31)
  const int so = (tid & 7) * 16;           // byte offset within the 128 B expert chunk
  const unsigned long long wbBase = (unsigned long long)Wb;
  const unsigned ldsBase = (unsigned)(unsigned long long)&wlds[0][0][0];

  // prologue: issue async loads for stage kc=0 into buffer 0
  {
    const unsigned long long g = wbBase + (unsigned long long)e0 * (D_DIM * 2) + so;
    async_b128_to_lds(ldsBase + e0 * (WPAD * 2) + so, g);
    async_b128_to_lds(ldsBase + (e0 + 32) * (WPAD * 2) + so, g + 32ull * (D_DIM * 2));
  }

  v8f c0 = {}, c1 = {}, c2 = {}, c3 = {};
  int buf = 0;

  for (int kc = 0; kc < D_DIM; kc += KC) {
    // issue next stage's async loads into the other buffer (overlaps with this stage's WMMAs)
    if (kc + KC < D_DIM) {
      const unsigned nb = ldsBase + (buf ^ 1) * (E_DIM * WPAD * 2);
      const unsigned long long g =
          wbBase + (unsigned long long)e0 * (D_DIM * 2) + (kc + KC) * 2 + so;
      async_b128_to_lds(nb + e0 * (WPAD * 2) + so, g);
      async_b128_to_lds(nb + (e0 + 32) * (WPAD * 2) + so, g + 32ull * (D_DIM * 2));
      asm volatile("s_wait_asynccnt 0x2" ::: "memory");   // current buffer's 2 loads done
    } else {
      asm volatile("s_wait_asynccnt 0x0" ::: "memory");
    }
    __syncthreads();   // all waves' async writes for `buf` visible

    if (kc + KC < D_DIM) __builtin_prefetch(xrow + kc + KC, 0, 0); // global_prefetch_b8 for x

#pragma unroll
    for (int kk = 0; kk < KC; kk += 32) {
      // ---- B fragments first (batch the ds loads; one dscnt wait covers 4 WMMAs)
      const int kb = kk + hi * 16;
      v16bf b0 = *(const v16bf*)&wlds[buf][laneLo +  0][kb];
      v16bf b1 = *(const v16bf*)&wlds[buf][laneLo + 16][kb];
      v16bf b2 = *(const v16bf*)&wlds[buf][laneLo + 32][kb];
      v16bf b3 = *(const v16bf*)&wlds[buf][laneLo + 48][kb];

      // ---- A fragment (16x32 bf16): lanes 0-15 K=[0..15]; lanes 16-31 K=[16..31]
      const float* ap = xrow + kc + kk + hi * 16;
      float4 f0 = *(const float4*)(ap);
      float4 f1 = *(const float4*)(ap + 4);
      float4 f2 = *(const float4*)(ap + 8);
      float4 f3 = *(const float4*)(ap + 12);
      v16bf a;
      a[0]  = to_bf16(f0.x); a[1]  = to_bf16(f0.y); a[2]  = to_bf16(f0.z); a[3]  = to_bf16(f0.w);
      a[4]  = to_bf16(f1.x); a[5]  = to_bf16(f1.y); a[6]  = to_bf16(f1.z); a[7]  = to_bf16(f1.w);
      a[8]  = to_bf16(f2.x); a[9]  = to_bf16(f2.y); a[10] = to_bf16(f2.z); a[11] = to_bf16(f2.w);
      a[12] = to_bf16(f3.x); a[13] = to_bf16(f3.y); a[14] = to_bf16(f3.z); a[15] = to_bf16(f3.w);

      c0 = __builtin_amdgcn_wmma_f32_16x16x32_bf16(false, a, false, b0, (short)0, c0, false, false);
      c1 = __builtin_amdgcn_wmma_f32_16x16x32_bf16(false, a, false, b1, (short)0, c1, false, false);
      c2 = __builtin_amdgcn_wmma_f32_16x16x32_bf16(false, a, false, b2, (short)0, c2, false, false);
      c3 = __builtin_amdgcn_wmma_f32_16x16x32_bf16(false, a, false, b3, (short)0, c3, false, false);
    }

    __syncthreads();   // all waves done reading `buf` before it gets overwritten
    buf ^= 1;
  }

  // ---- spill logits to LDS per the 16x16 f32 C layout (VGPR j: lanes0-15 M=j, lanes16-31 M=j+8)
#pragma unroll
  for (int j = 0; j < 8; ++j) {
    const int rl = wave * 16 + j + hi * 8;
    sh_logits[rl][laneLo +  0] = c0[j];
    sh_logits[rl][laneLo + 16] = c1[j];
    sh_logits[rl][laneLo + 32] = c2[j];
    sh_logits[rl][laneLo + 48] = c3[j];
  }
  __syncthreads();

  // ---- phase 2: one thread per token row
  if (tid < ROWS_PER_BLOCK) {
    const int n = rowBase + tid;
    float best = -1e30f, sec = -1e30f;
    int bi = 0, si = 0;
    for (int e = 0; e < E_DIM; ++e) {          // ascending: first index wins on ties
      float v = sh_logits[tid][e];
      if (v > best) { sec = best; si = bi; best = v; bi = e; }
      else if (v > sec) { sec = v; si = e; }
    }
    float sum = 0.0f;
    for (int e = 0; e < E_DIM; ++e) sum += __expf(sh_logits[tid][e] - best);
    const float inv = 1.0f / sum;
    for (int i = 0; i < E_DIM; ++i) {          // staggered to spread LDS atomic banks
      const int e = (i + tid) & (E_DIM - 1);
      atomicAdd(&sh_acc[e], __expf(sh_logits[tid][e] - best) * inv);
    }
    out[2 * n + 0] = (float)bi;
    out[2 * n + 1] = (float)si;
    const float p0 = 1.0f / (1.0f + __expf(sec - best));
    out[2 * N + 2 * n + 0] = p0;
    out[2 * N + 2 * n + 1] = 1.0f - p0;
  }
  __syncthreads();
  if (tid < E_DIM) atomicAdd(&expert_acc[tid], sh_acc[tid]);
}

__global__ __launch_bounds__(64) void moe_gate_cv(const float* __restrict__ acc,
                                                  float* __restrict__ out, int N) {
  __shared__ float m[E_DIM];
  const int e = threadIdx.x;
  m[e] = acc[e] / (float)N;
  __syncthreads();
  if (e == 0) {
    float mu = 0.0f;
    for (int i = 0; i < E_DIM; ++i) mu += m[i];
    mu *= (1.0f / E_DIM);
    float ss = 0.0f;
    for (int i = 0; i < E_DIM; ++i) { float d = m[i] - mu; ss += d * d; }
    const float sd = sqrtf(ss / (float)(E_DIM - 1));
    const float cv = sd / (mu + 1e-10f);
    out[4 * N] = cv * cv;
  }
}

extern "C" void kernel_launch(void* const* d_in, const int* in_sizes, int n_in,
                              void* d_out, int out_size, void* d_ws, size_t ws_size,
                              hipStream_t stream) {
  const float* x = (const float*)d_in[0];
  const float* W = (const float*)d_in[1];
  float* out = (float*)d_out;
  float* acc  = (float*)d_ws;                            // 64 floats
  __bf16* Wb  = (__bf16*)((char*)d_ws + 256);            // 512 KB bf16 copy of W
  const int N = in_sizes[0] / D_DIM;                     // B*S = 16384
  const int wElems = in_sizes[1];                        // E*D = 262144

  moe_gate_prep<<<wElems / (256 * 8), 256, 0, stream>>>(W, Wb, acc);
  moe_gate_gemm<<<N / ROWS_PER_BLOCK, 256, 0, stream>>>(x, Wb, out, acc, N);
  moe_gate_cv<<<1, E_DIM, 0, stream>>>(acc, out, N);
}
